// TemporalGNN_61375082660331
// MI455X (gfx1250) — matrix operand order repaired and checked by reading, not compiled, not run on hardware
//
#include <hip/hip_runtime.h>

// ---------------------------------------------------------------------------
// TemporalGNN (2x TransformerConv, heads=1) for gfx1250 / MI455X
//   - dense projections via v_wmma_f32_16x16x32_f16 (f32 accumulate)
//   - dual accumulators interleave WMMAs to hide the D->A/B hazard slots
//   - bias folded into WMMA accumulator, skip written straight to output
//   - edge phase: one wave32 per edge, float4 per lane, shfl_xor reduction,
//     f32 global atomics for segment max / sum / scatter-add
// ---------------------------------------------------------------------------

typedef __attribute__((ext_vector_type(16))) _Float16 v16h;
typedef __attribute__((ext_vector_type(8)))  float    v8f;

#define N_CH 128   // IN_C == HID_C == OUT_C == 128

__device__ __forceinline__ void atomicMaxF(float* addr, float val) {
  // standard monotonic bit trick: works across mixed signs with -inf init
  if (val >= 0.0f) atomicMax((int*)addr, __float_as_int(val));
  else             atomicMin((unsigned int*)addr, __float_as_uint(val));
}

// ---------------------------------------------------------------------------
// Pre-pack all 8 weight matrices (f32 [K=128][N=128]) into per-lane WMMA
// B-fragments (f16).  Fragment layout packed here (contiguous k) for V_WMMA_F32_16X16X32_F16 B (32x16):
//   lane <16 : column N = lane,    halves 0..15 = K  0..15 of the K-block
//   lane>=16 : column N = lane-16, halves 0..15 = K 16..31 of the K-block
// Linear layout: frag[(m*8 + nt)*4 + kb][lane][16 halves]
// ---------------------------------------------------------------------------
__global__ __launch_bounds__(256) void prep_frags(
    const float* w0, const float* w1, const float* w2, const float* w3,
    const float* w4, const float* w5, const float* w6, const float* w7,
    _Float16* __restrict__ frags)
{
  int t = blockIdx.x * blockDim.x + threadIdx.x;
  if (t >= 8 * 8 * 4 * 32) return;
  int lane = t & 31;
  int grp  = t >> 5;            // == (m*8 + nt)*4 + kb
  int kb   = grp & 3;
  int nt   = (grp >> 2) & 7;
  int m    = grp >> 5;
  const float* W;
  switch (m) {
    case 0: W = w0; break; case 1: W = w1; break;
    case 2: W = w2; break; case 3: W = w3; break;
    case 4: W = w4; break; case 5: W = w5; break;
    case 6: W = w6; break; default: W = w7; break;
  }
  int ncol  = nt * 16 + (lane & 15);
  int kbase = kb * 32 + ((lane < 16) ? 0 : 16);
  _Float16* dstp = frags + ((size_t)grp * 32 + lane) * 16;
#pragma unroll
  for (int hh = 0; hh < 16; ++hh)
    dstp[hh] = (_Float16)W[(size_t)(kbase + hh) * N_CH + ncol];
}

// ---------------------------------------------------------------------------
// Fused Q/K/V/Skip projection: one block = 16 rows of x, 8 waves.
// Wave w: output matrix om = w>>1 (Q,K,V,S), column half = w&1 (4 n-tiles,
// processed as 2 pairs with independent accumulators).
// A-fragment (16x32 f16), contiguous k order matching the B frags:
//   lane <16 : row M = lane,    halves = K 0..15 of the K-block
//   lane>=16 : row M = lane-16, halves = K 16..31
// Accumulator initialized with bias (exact f32); optional ReLU on input.
// ---------------------------------------------------------------------------
__global__ __launch_bounds__(256) void gemm_qkvs(
    const float* __restrict__ x, const _Float16* __restrict__ fragsL,
    const float* __restrict__ b0, const float* __restrict__ b1,
    const float* __restrict__ b2, const float* __restrict__ b3,
    float* __restrict__ o0, float* __restrict__ o1,
    float* __restrict__ o2, float* __restrict__ o3,
    int nN, int relu)
{
  const int lane = threadIdx.x & 31;
  const int wave = threadIdx.x >> 5;
  const int row0 = blockIdx.x * 16;
  if (row0 >= nN) return;

  const int om    = wave >> 1;   // 0..3 -> Q,K,V,S
  const int nhalf = wave & 1;    // column half
  const float* bias = (om == 0) ? b0 : (om == 1) ? b1 : (om == 2) ? b2 : b3;
  float*       out  = (om == 0) ? o0 : (om == 1) ? o1 : (om == 2) ? o2 : o3;

  int arow = row0 + (lane & 15);
  if (arow >= nN) arow = nN - 1;              // clamp (partial-tile fallback)
  const float* xr  = x + (size_t)arow * N_CH;
  const int   kofs = (lane < 16) ? 0 : 16;

  v16h a[4];
#pragma unroll
  for (int kb = 0; kb < 4; ++kb) {
    const int kbase = kb * 32 + kofs;
#pragma unroll
    for (int h = 0; h < 8; ++h) {
      float f0 = xr[kbase + h];
      float f1 = xr[kbase + 8 + h];
      if (relu) { f0 = fmaxf(f0, 0.0f); f1 = fmaxf(f1, 0.0f); }
      a[kb][h]     = (_Float16)f0;
      a[kb][8 + h] = (_Float16)f1;
    }
  }

  // B-fragment base for this wave's output matrix, pre-offset by lane
  const v16h* bf = (const v16h*)fragsL + (size_t)om * 8 * 4 * 32 + lane;
  const bool fullTile = (row0 + 16 <= nN);
  const int  mrow = row0 + ((lane < 16) ? 0 : 8);

#pragma unroll
  for (int p = 0; p < 2; ++p) {
    const int nt0 = nhalf * 4 + p * 2;      // pair of n-tiles
    const int nc0 = nt0 * 16 + (lane & 15);
    const int nc1 = nc0 + 16;
    const float bv0 = bias[nc0];
    const float bv1 = bias[nc1];
    v8f c0 = { bv0, bv0, bv0, bv0, bv0, bv0, bv0, bv0 };
    v8f c1 = { bv1, bv1, bv1, bv1, bv1, bv1, bv1, bv1 };
#pragma unroll
    for (int kb = 0; kb < 4; ++kb) {
      v16h bfr0 = bf[((size_t)nt0 * 4 + kb) * 32];
      v16h bfr1 = bf[((size_t)(nt0 + 1) * 4 + kb) * 32];
      // interleaved independent accumulators -> no dead hazard slots
      c0 = __builtin_amdgcn_wmma_f32_16x16x32_f16(false, a[kb], false, bfr0,
                                                  (short)0, c0, false, false);
      c1 = __builtin_amdgcn_wmma_f32_16x16x32_f16(false, a[kb], false, bfr1,
                                                  (short)0, c1, false, false);
    }
    if (fullTile) {
#pragma unroll
      for (int j = 0; j < 8; ++j) {
        out[(size_t)(mrow + j) * N_CH + nc0] = c0[j];
        out[(size_t)(mrow + j) * N_CH + nc1] = c1[j];
      }
    } else {
#pragma unroll
      for (int j = 0; j < 8; ++j) {
        int r = mrow + j;
        if (r < nN) {
          out[(size_t)r * N_CH + nc0] = c0[j];
          out[(size_t)r * N_CH + nc1] = c1[j];
        }
      }
    }
  }
}

__global__ __launch_bounds__(256) void init_seg(float* __restrict__ segmax,
                                                float* __restrict__ segsum, int n)
{
  int i = blockIdx.x * blockDim.x + threadIdx.x;
  if (i < n) {
    segmax[i] = __uint_as_float(0xFF800000u);  // -inf
    segsum[i] = 0.0f;
  }
}

// score[e] = <q[dst], k[src]> / sqrt(128); segment max over dst via atomics.
__global__ __launch_bounds__(256) void edge_score(
    const float* __restrict__ q, const float* __restrict__ k,
    const int* __restrict__ src, const int* __restrict__ dst,
    float* __restrict__ score, float* __restrict__ segmax, int nE)
{
  int wid  = (blockIdx.x * blockDim.x + threadIdx.x) >> 5;
  int lane = threadIdx.x & 31;
  if (wid >= nE) return;
  int s = src[wid], d = dst[wid];
  float4 qv = *((const float4*)(q + (size_t)d * N_CH) + lane);
  float4 kv = *((const float4*)(k + (size_t)s * N_CH) + lane);
  float acc = qv.x * kv.x + qv.y * kv.y + qv.z * kv.z + qv.w * kv.w;
#pragma unroll
  for (int off = 16; off > 0; off >>= 1)
    acc += __shfl_xor(acc, off, 32);
  if (lane == 0) {
    float sc = acc * 0.08838834764831845f;    // 1/sqrt(128)
    score[wid] = sc;
    atomicMaxF(segmax + d, sc);
  }
}

// ex[e] = exp(score - segmax[dst]); segment sum over dst.
__global__ __launch_bounds__(256) void edge_expsum(
    const float* __restrict__ score, const int* __restrict__ dst,
    const float* __restrict__ segmax, float* __restrict__ ex,
    float* __restrict__ segsum, int nE)
{
  int e = blockIdx.x * blockDim.x + threadIdx.x;
  if (e >= nE) return;
  int d = dst[e];
  float v = __expf(score[e] - segmax[d]);
  ex[e] = v;
  atomicAdd(segsum + d, v);
}

// out[dst] += (ex/segsum[dst]) * v[src]   (out already holds the skip term)
__global__ __launch_bounds__(256) void edge_scatter(
    const float* __restrict__ vmat, const float* __restrict__ ex,
    const float* __restrict__ segsum, const int* __restrict__ src,
    const int* __restrict__ dst, float* __restrict__ out, int nE)
{
  int wid  = (blockIdx.x * blockDim.x + threadIdx.x) >> 5;
  int lane = threadIdx.x & 31;
  if (wid >= nE) return;
  int s = src[wid], d = dst[wid];
  float alpha = ex[wid] / segsum[d];
  float4 vv = *((const float4*)(vmat + (size_t)s * N_CH) + lane);
  float* o = out + (size_t)d * N_CH + lane * 4;
  atomicAdd(o + 0, alpha * vv.x);
  atomicAdd(o + 1, alpha * vv.y);
  atomicAdd(o + 2, alpha * vv.z);
  atomicAdd(o + 3, alpha * vv.w);
}

// ---------------------------------------------------------------------------
extern "C" void kernel_launch(void* const* d_in, const int* in_sizes, int n_in,
                              void* d_out, int out_size, void* d_ws, size_t ws_size,
                              hipStream_t stream)
{
  const float* x  = (const float*)d_in[0];
  const int*   ei = (const int*)d_in[1];
  const int nN = in_sizes[0] / N_CH;
  const int nE = in_sizes[1] / 2;
  const int* src = ei;
  const int* dst = ei + nE;

  const float* W[8]; const float* B[8];
  for (int i = 0; i < 8; ++i) {
    W[i] = (const float*)d_in[2 + 2 * i];
    B[i] = (const float*)d_in[3 + 2 * i];
  }

  // workspace carve-out (floats)
  const size_t NC = (size_t)nN * N_CH;
  float* q      = (float*)d_ws;
  float* k      = q + NC;
  float* v      = k + NC;
  float* h      = v + NC;          // layer-1 output (relu fused into layer-2 load)
  float* score  = h + NC;
  float* ex     = score + nE;
  float* segmax = ex + nE;
  float* segsum = segmax + nN;
  _Float16* frags = (_Float16*)(segsum + nN);   // 8*8*4*32*16 halves = 256 KB
  (void)ws_size; (void)n_in;

  const int rowBlocks      = (nN + 15) / 16;
  const int edgeWaveBlocks = (nE + 7) / 8;      // one wave32 per edge
  const int nodeBlocks     = (nN + 255) / 256;
  const int edgeBlocks     = (nE + 255) / 256;
  const size_t FRAGS_PER_LAYER = (size_t)4 * 8 * 4 * 32 * 16;  // halves

  prep_frags<<<(8 * 8 * 4 * 32 + 255) / 256, 256, 0, stream>>>(
      W[0], W[1], W[2], W[3], W[4], W[5], W[6], W[7], frags);

  // ---------------- layer 1: x -> h ----------------
  init_seg<<<nodeBlocks, 256, 0, stream>>>(segmax, segsum, nN);
  gemm_qkvs<<<rowBlocks, 256, 0, stream>>>(x, frags,
      B[0], B[1], B[2], B[3], q, k, v, h, nN, /*relu=*/0);
  edge_score<<<edgeWaveBlocks, 256, 0, stream>>>(q, k, src, dst, score, segmax, nE);
  edge_expsum<<<edgeBlocks, 256, 0, stream>>>(score, dst, segmax, ex, segsum, nE);
  edge_scatter<<<edgeWaveBlocks, 256, 0, stream>>>(v, ex, segsum, src, dst, h, nE);

  // ---------------- layer 2: relu(h) -> out ----------------
  float* outp = (float*)d_out;
  init_seg<<<nodeBlocks, 256, 0, stream>>>(segmax, segsum, nN);
  gemm_qkvs<<<rowBlocks, 256, 0, stream>>>(h, frags + FRAGS_PER_LAYER,
      B[4], B[5], B[6], B[7], q, k, v, outp, nN, /*relu=*/1);
  edge_score<<<edgeWaveBlocks, 256, 0, stream>>>(q, k, src, dst, score, segmax, nE);
  edge_expsum<<<edgeBlocks, 256, 0, stream>>>(score, dst, segmax, ex, segsum, nE);
  edge_scatter<<<edgeWaveBlocks, 256, 0, stream>>>(v, ex, segsum, src, dst, outp, nE);
  (void)out_size;
}
